// SwinTransformerBlock3D_32100585570776
// MI455X (gfx1250) — matrix operand order repaired and checked by reading, not compiled, not run on hardware
//
#include <hip/hip_runtime.h>
#include <math.h>

typedef __attribute__((ext_vector_type(16))) _Float16 v16h;
typedef __attribute__((ext_vector_type(8)))  float    v8f;

static __device__ __forceinline__ v8f wmma_f16(v16h a, v16h b, v8f c) {
  // (neg_a, A, neg_b, B, c_mod, C, reuse_a, reuse_b)
  return __builtin_amdgcn_wmma_f32_16x16x32_f16(false, a, false, b, (short)0, c, false, false);
}
static __device__ __forceinline__ v16h ldv(const _Float16* p) { return *(const v16h*)p; }

// Async global->LDS copy, 16B per lane (tracked by ASYNCcnt; no VGPR round trip).
// LDS operand = byte offset within workgroup LDS (low 32 bits of generic shared addr).
static __device__ __forceinline__ void cp_async128(void* lds, const void* g) {
  unsigned l = (unsigned)(uintptr_t)lds;
  asm volatile("global_load_async_to_lds_b128 %0, %1, off"
               :: "v"(l), "v"(g) : "memory");
}
static __device__ __forceinline__ void wait_async() {
  asm volatile("s_wait_asynccnt 0x0" ::: "memory");
}

// Half-offset of element (row m, col c) inside a 16-row A-tile stored in
// fragment order [kt][lane][16]  (contiguous k order, same as the B packing):
//   frag element (lane,i) holds k = ((lane>>4)<<4) | i, m = lane&15
static __device__ __forceinline__ int aswz(int m, int c) {
  int kt = c >> 5, kl = c & 31;
  return (kt * 32 + ((kl >> 4) << 4) + m) * 16 + (kl & 15);
}

// shifted-window token -> flat global token (same map for gather in K1 and scatter in K4)
static __device__ __forceinline__ int win_token_to_global(int wi, int t) {
  int b = wi >> 8; int r = wi & 255;
  int zd = r >> 6; r &= 63; int zh = r >> 3; int zw = r & 7;
  int id = t / 49; int rr = t - id * 49; int ih = rr / 7; int iw = rr - ih * 7;
  int ds = (zd * 2 + id + 1) & 7;
  int hs = zh * 7 + ih + 3; if (hs >= 56) hs -= 56;
  int ws = zw * 7 + iw + 3; if (ws >= 56) ws -= 56;
  return ((b * 8 + ds) * 56 + hs) * 56 + ws;
}

// ---- K0: weights fp32 -> f16 in B-fragment order [kt][nt][lane][16] ----
static __device__ __forceinline__ void wswz(const float* __restrict__ W,
                                            _Float16* __restrict__ dst,
                                            int d, int N, int NT) {
  int i2 = d & 15, lane = (d >> 4) & 31, tile = d >> 9;
  int nt = tile % NT, kt = tile / NT;
  int k = kt * 32 + ((lane >> 4) << 4) + i2;   // B frag: k = (lane>>4)*16 + i
  int n = nt * 16 + (lane & 15);
  dst[d] = (_Float16)W[k * N + n];
}
__global__ void k0_cvt_w(const float* __restrict__ qkvw, const float* __restrict__ projw,
                         const float* __restrict__ fc1w, const float* __restrict__ fc2w,
                         _Float16* __restrict__ qh, _Float16* __restrict__ ph,
                         _Float16* __restrict__ f1h, _Float16* __restrict__ f2h) {
  int i = blockIdx.x * 256 + threadIdx.x;
  if (i < 49152)       wswz(qkvw, qh, i, 384, 24);
  else if (i < 65536)  wswz(projw, ph, i - 49152, 128, 8);
  else if (i < 131072) wswz(fc1w, f1h, i - 65536, 512, 32);
  else if (i < 196608) wswz(fc2w, f2h, i - 131072, 128, 8);
}

// ---- K1: LN1 + shift + partition -> xw f16, A-swizzled [1024][7][4*512] ----
__global__ __launch_bounds__(128) void k1_ln_shift_part(
    const float* __restrict__ x, const float* __restrict__ w,
    const float* __restrict__ bch, _Float16* __restrict__ xw) {
  int row = blockIdx.x;                       // [0, 1024*112)
  int wi = row / 112, t = row - wi * 112;
  int c = threadIdx.x;                        // 128 threads = 1 channel each
  int tile = t >> 4, m = t & 15;
  size_t dst = (size_t)(wi * 7 + tile) * 2048 + aswz(m, c);
  __shared__ float2 red[128];
  if (t >= 98) { xw[dst] = (_Float16)0.f; return; }
  int g = win_token_to_global(wi, t);
  float v = x[(size_t)g * 128 + c];
  red[c] = make_float2(v, v * v);
  __syncthreads();
  for (int s = 64; s > 0; s >>= 1) {
    if (c < s) { red[c].x += red[c + s].x; red[c].y += red[c + s].y; }
    __syncthreads();
  }
  float mn  = red[0].x * (1.f / 128.f);
  float var = red[0].y * (1.f / 128.f) - mn * mn;
  float inv = rsqrtf(var + 1e-5f);
  xw[dst] = (_Float16)((v - mn) * inv * w[c] + bch[c]);
}

// ---- K2: QKV GEMM; q->A-swizzled, k->B^T-swizzled, v->B-swizzled (per head) ----
__global__ __launch_bounds__(256) void k2_qkv(
    const _Float16* __restrict__ xw, const _Float16* __restrict__ wq,
    const float* __restrict__ bias,
    _Float16* __restrict__ qsw, _Float16* __restrict__ ksw, _Float16* __restrict__ vsw) {
  int wi = blockIdx.x / 7, mt = blockIdx.x % 7;
  __shared__ __align__(32) _Float16 As[2048];
  cp_async128(As + threadIdx.x * 8, xw + (size_t)(wi * 7 + mt) * 2048 + threadIdx.x * 8);
  wait_async();
  __syncthreads();
  int wave = threadIdx.x >> 5, lane = threadIdx.x & 31;
#pragma unroll
  for (int j = 0; j < 3; ++j) {               // 8 waves * 3 = 24 n-tiles = 384 cols
    int nt = wave * 3 + j;
    v8f acc = {};
#pragma unroll
    for (int ks = 0; ks < 4; ++ks) {
      v16h a = ldv(As + (ks * 32 + lane) * 16);
      v16h b = ldv(wq + ((size_t)(ks * 24 + nt) * 32 + lane) * 16);
      acc = wmma_f16(a, b, acc);
    }
    int n = nt * 16 + (lane & 15);
    float bs = bias[n];
    float scl = (nt < 8) ? 0.17677669529663687f : 1.0f;   // 1/sqrt(32) folded into q
    int hh = (n & 127) >> 5, c32 = n & 31;
#pragma unroll
    for (int r = 0; r < 8; ++r) {
      int rr = mt * 16 + ((lane >> 4) << 3) + r;
      int m = rr & 15;
      _Float16 val = (_Float16)((acc[r] + bs) * scl);
      if (nt < 8) {                                        // Q: A-swizzled per (tile, head)
        qsw[(size_t)((wi * 7 + mt) * 4 + hh) * 512 + aswz(m, c32)] = val;
      } else if (nt < 16) {                                // K: B^T frag order per head
        int lane2 = ((c32 >> 4) << 4) + m;                 // n=token, k=channel
        ksw[((size_t)((wi * 4 + hh) * 7 + mt) * 32 + lane2) * 16 + (c32 & 15)] = val;
      } else {                                             // V: B frag order, k=token pad->128
        int ks2 = rr >> 5, tl = rr & 31;
        int lane2 = ((tl >> 4) << 4) + (c32 & 15);         // n=channel, k=token
        vsw[((size_t)(((wi * 4 + hh) * 4 + ks2) * 2 + (c32 >> 4)) * 32 + lane2) * 16 + (tl & 15)] = val;
      }
    }
  }
}

// ---- K3: attention per (window, head); all fragments = contiguous 32B LDS loads ----
__global__ __launch_bounds__(256) void k3_attn(
    const _Float16* __restrict__ qsw, const _Float16* __restrict__ ksw,
    const _Float16* __restrict__ vsw, _Float16* __restrict__ aosw) {
  int wi = blockIdx.x >> 2, hh = blockIdx.x & 3;
  __shared__ __align__(32) _Float16 Qs[3584];   // [7 tiles][32][16]
  __shared__ __align__(32) _Float16 Ks[3584];   // [7 ntiles][32][16]
  __shared__ __align__(32) _Float16 Vs[4096];   // [4 ks][2 nt2][32][16]
  __shared__ __align__(32) _Float16 Ss[14336];  // [7 tiles][4 kt][32][16] probs
  for (int c = threadIdx.x; c < 448; c += 256) {            // Q: per-tile blocks (stride = head)
    int t = c >> 6, w64 = c & 63;
    cp_async128(Qs + c * 8, qsw + (size_t)((wi * 7 + t) * 4 + hh) * 512 + w64 * 8);
  }
  for (int c = threadIdx.x; c < 448; c += 256)
    cp_async128(Ks + c * 8, ksw + (size_t)(wi * 4 + hh) * 3584 + c * 8);
  for (int c = threadIdx.x; c < 512; c += 256)
    cp_async128(Vs + c * 8, vsw + (size_t)(wi * 4 + hh) * 4096 + c * 8);
  wait_async();
  __syncthreads();
  int wave = threadIdx.x >> 5, lane = threadIdx.x & 31;
  int n15 = lane & 15, g8 = (lane >> 4) << 3;
  if (wave < 7) {                             // S = Q K^T (wave-uniform branch)
    v16h a = ldv(Qs + (wave * 32 + lane) * 16);
#pragma unroll
    for (int nt = 0; nt < 7; ++nt) {
      v8f acc = {};
      v16h b = ldv(Ks + (nt * 32 + lane) * 16);
      acc = wmma_f16(a, b, acc);
#pragma unroll
      for (int r = 0; r < 8; ++r)
        Ss[wave * 2048 + aswz(g8 + r, nt * 16 + n15)] = (_Float16)acc[r];
    }
  }
  __syncthreads();
  if (threadIdx.x < 112) {                    // masked softmax over 98 valid cols
    int tile = threadIdx.x >> 4, m = threadIdx.x & 15;
    _Float16* base = Ss + tile * 2048;
    float mx = -1e30f;
    for (int c = 0; c < 98; ++c) mx = fmaxf(mx, (float)base[aswz(m, c)]);
    float sum = 0.f;
    for (int c = 0; c < 98; ++c) {
      int o = aswz(m, c);
      float e = __expf((float)base[o] - mx);
      sum += e; base[o] = (_Float16)e;
    }
    float inv = 1.f / sum;
    for (int c = 0; c < 98; ++c) {
      int o = aswz(m, c);
      base[o] = (_Float16)((float)base[o] * inv);
    }
    for (int c = 98; c < 128; ++c) base[aswz(m, c)] = (_Float16)0.f;  // zero pad K-dim
  }
  __syncthreads();
  if (wave < 7) {                             // O = P V
#pragma unroll
    for (int nt2 = 0; nt2 < 2; ++nt2) {
      v8f acc = {};
#pragma unroll
      for (int ks = 0; ks < 4; ++ks) {
        v16h a = ldv(Ss + wave * 2048 + (ks * 32 + lane) * 16);
        v16h b = ldv(Vs + ((ks * 2 + nt2) * 32 + lane) * 16);
        acc = wmma_f16(a, b, acc);
      }
#pragma unroll
      for (int r = 0; r < 8; ++r)
        aosw[(size_t)(wi * 7 + wave) * 2048 + aswz(g8 + r, hh * 32 + nt2 * 16 + n15)] =
            (_Float16)acc[r];
    }
  }
}

// ---- K4: proj GEMM + window reverse + residual -> y fp32 ----
__global__ __launch_bounds__(256) void k4_proj_res(
    const _Float16* __restrict__ aosw, const _Float16* __restrict__ pw,
    const float* __restrict__ pb, const float* __restrict__ x,
    float* __restrict__ y) {
  int wi = blockIdx.x / 7, mt = blockIdx.x % 7;
  __shared__ __align__(32) _Float16 As[2048];
  cp_async128(As + threadIdx.x * 8, aosw + (size_t)(wi * 7 + mt) * 2048 + threadIdx.x * 8);
  wait_async();
  __syncthreads();
  int wave = threadIdx.x >> 5, lane = threadIdx.x & 31;
  v8f acc = {};
#pragma unroll
  for (int ks = 0; ks < 4; ++ks) {
    v16h a = ldv(As + (ks * 32 + lane) * 16);
    v16h b = ldv(pw + ((size_t)(ks * 8 + wave) * 32 + lane) * 16);
    acc = wmma_f16(a, b, acc);
  }
  int n = wave * 16 + (lane & 15);
  float bs = pb[n];
#pragma unroll
  for (int r = 0; r < 8; ++r) {
    int t = mt * 16 + ((lane >> 4) << 3) + r;
    if (t < 98) {
      int g = win_token_to_global(wi, t);
      y[(size_t)g * 128 + n] = x[(size_t)g * 128 + n] + acc[r] + bs;
    }
  }
}

// ---- K5: LN2 (fused) + fc1 GEMM + SiLU -> h1 f16 (A-swizzled [mt][16*512]) ----
__global__ __launch_bounds__(256) void k5_ln2_fc1(
    const float* __restrict__ y, const float* __restrict__ w2,
    const float* __restrict__ b2, const _Float16* __restrict__ fw,
    const float* __restrict__ fb, _Float16* __restrict__ h1) {
  int mt = blockIdx.x;
  __shared__ __align__(16) float Ys[2048];
  __shared__ __align__(32) _Float16 As[2048];
  __shared__ float mean_s[16], inv_s[16];
  for (int c = threadIdx.x; c < 512; c += 256)
    cp_async128(Ys + c * 4, y + (size_t)mt * 2048 + c * 4);
  wait_async();
  __syncthreads();
  if (threadIdx.x < 16) {
    float s = 0.f, s2 = 0.f;
    for (int c = 0; c < 128; ++c) { float v = Ys[threadIdx.x * 128 + c]; s += v; s2 += v * v; }
    float m = s * (1.f / 128.f);
    mean_s[threadIdx.x] = m;
    inv_s[threadIdx.x] = rsqrtf(s2 * (1.f / 128.f) - m * m + 1e-5f);
  }
  __syncthreads();
  for (int e = threadIdx.x; e < 2048; e += 256) {
    int r = e >> 7, c = e & 127;
    As[aswz(r, c)] = (_Float16)((Ys[e] - mean_s[r]) * inv_s[r] * w2[c] + b2[c]);
  }
  __syncthreads();
  int wave = threadIdx.x >> 5, lane = threadIdx.x & 31;
#pragma unroll
  for (int j = 0; j < 4; ++j) {               // 8 waves * 4 = 32 n-tiles = 512 cols
    int nt = wave * 4 + j;
    v8f acc = {};
#pragma unroll
    for (int ks = 0; ks < 4; ++ks) {
      v16h a = ldv(As + (ks * 32 + lane) * 16);
      v16h b = ldv(fw + ((size_t)(ks * 32 + nt) * 32 + lane) * 16);
      acc = wmma_f16(a, b, acc);
    }
    int n = nt * 16 + (lane & 15);
    float bs = fb[n];
#pragma unroll
    for (int r = 0; r < 8; ++r) {
      int m = ((lane >> 4) << 3) + r;
      float v = acc[r] + bs;
      h1[(size_t)mt * 8192 + aswz(m, n)] = (_Float16)(v / (1.f + __expf(-v)));  // SiLU
    }
  }
}

// ---- K6: fc2 GEMM + residual -> out fp32 ----
__global__ __launch_bounds__(256) void k6_fc2_res(
    const _Float16* __restrict__ h1, const _Float16* __restrict__ fw,
    const float* __restrict__ fb, const float* __restrict__ y,
    float* __restrict__ out) {
  int mt = blockIdx.x;
  __shared__ __align__(32) _Float16 Hs[8192];  // 16 KB, [16 kt][32][16]
  for (int c = threadIdx.x; c < 1024; c += 256)
    cp_async128(Hs + c * 8, h1 + (size_t)mt * 8192 + c * 8);
  wait_async();
  __syncthreads();
  int wave = threadIdx.x >> 5, lane = threadIdx.x & 31;
  v8f acc = {};
#pragma unroll
  for (int ks = 0; ks < 16; ++ks) {
    v16h a = ldv(Hs + (ks * 32 + lane) * 16);
    v16h b = ldv(fw + ((size_t)(ks * 8 + wave) * 32 + lane) * 16);
    acc = wmma_f16(a, b, acc);
  }
  int n = wave * 16 + (lane & 15);
  float bs = fb[n];
#pragma unroll
  for (int r = 0; r < 8; ++r) {
    int rr = mt * 16 + ((lane >> 4) << 3) + r;
    out[(size_t)rr * 128 + n] = y[(size_t)rr * 128 + n] + acc[r] + bs;
  }
}

extern "C" void kernel_launch(void* const* d_in, const int* in_sizes, int n_in,
                              void* d_out, int out_size, void* d_ws, size_t ws_size,
                              hipStream_t stream) {
  (void)in_sizes; (void)n_in; (void)out_size; (void)ws_size;
  const float* x     = (const float*)d_in[0];
  const float* n1w   = (const float*)d_in[1];
  const float* n1b   = (const float*)d_in[2];
  const float* qkv_w = (const float*)d_in[3];
  const float* qkv_b = (const float*)d_in[4];
  const float* prj_w = (const float*)d_in[5];
  const float* prj_b = (const float*)d_in[6];
  const float* n2w   = (const float*)d_in[7];
  const float* n2b   = (const float*)d_in[8];
  const float* fc1_w = (const float*)d_in[9];
  const float* fc1_b = (const float*)d_in[10];
  const float* fc2_w = (const float*)d_in[11];
  const float* fc2_b = (const float*)d_in[12];

  // workspace layout (bytes), regions aliased across pipeline stages:
  //   [0,          29360128)   xw (A-swz) -> attn_out (A-swz) -> part of h1
  //   [29360128,  121634816)   q/k/v frag-swizzled -> rest of h1 (102760448 B total)
  //   [121634816, 173015040)   y fp32
  //   [173015040, 173408256)   f16 weights in B-fragment order
  char* ws = (char*)d_ws;
  _Float16* xw    = (_Float16*)(ws + 0);                 // also attn_out
  _Float16* qsw   = (_Float16*)(ws + 29360128);          // [1024][7][4][512]
  _Float16* ksw   = (_Float16*)(ws + 58720256);          // [1024][4][7][512]
  _Float16* vsw   = (_Float16*)(ws + 88080384);          // [1024][4][8][512] (tok pad 128)
  _Float16* h1    = (_Float16*)(ws + 0);                 // aliases dead xw+q+k
  float*    y     = (float*)   (ws + 121634816);
  _Float16* qkvwh = (_Float16*)(ws + 173015040);
  _Float16* prjwh = qkvwh + 49152;
  _Float16* fc1wh = prjwh + 16384;
  _Float16* fc2wh = fc1wh + 65536;

  k0_cvt_w<<<768, 256, 0, stream>>>(qkv_w, prj_w, fc1_w, fc2_w,
                                    qkvwh, prjwh, fc1wh, fc2wh);
  k1_ln_shift_part<<<1024 * 112, 128, 0, stream>>>(x, n1w, n1b, xw);
  k2_qkv<<<1024 * 7, 256, 0, stream>>>(xw, qkvwh, qkv_b, qsw, ksw, vsw);
  k3_attn<<<1024 * 4, 256, 0, stream>>>(qsw, ksw, vsw, xw /* attn_out reuses xw */);
  k4_proj_res<<<1024 * 7, 256, 0, stream>>>(xw, prjwh, prj_b, x, y);
  k5_ln2_fc1<<<6272, 256, 0, stream>>>(y, n2w, n2b, fc1wh, fc1_b, h1);
  k6_fc2_res<<<6272, 256, 0, stream>>>(h1, fc2wh, fc2_b, y, (float*)d_out);
}
